// physicsinformed3Dresnet_19026705121521
// MI455X (gfx1250) — matrix operand order repaired and checked by reading, not compiled, not run on hardware
//
#include <hip/hip_runtime.h>

typedef __attribute__((ext_vector_type(16))) _Float16 v16h;
typedef __attribute__((ext_vector_type(8)))  _Float16 v8h;
typedef __attribute__((ext_vector_type(8)))  float    v8f;

#define NDIM 64
#define N3   262144          // 64^3 voxels
#define CCH  32              // padded channel count (NDHWC)
#define TAPS 125             // 5x5x5
#define TAP_CHUNK 25         // one kd-slice staged in LDS at a time
#define CONV_BLOCK 512       // 16 waves of 32
#define WAVES_PER_BLOCK 16

// ---------------------------------------------------------------------------
// pack quat (fp32 NCDHW, 4 ch) -> fp16 NDHWC padded to 32 ch (zeros)
// ---------------------------------------------------------------------------
__global__ void pack_quat_kernel(const float* __restrict__ q,
                                 _Float16* __restrict__ act) {
  int idx = blockIdx.x * blockDim.x + threadIdx.x;   // over N3*CCH
  if (idx >= N3 * CCH) return;
  int c = idx & (CCH - 1);
  int v = idx >> 5;
  float x = (c < 4) ? q[c * N3 + v] : 0.f;
  act[idx] = (_Float16)x;
}

// ---------------------------------------------------------------------------
// Implicit-GEMM 5x5x5 circular conv, C=32 padded, via v_wmma_f32_16x16x32_f16.
// Each wave: M=16 consecutive voxels along w, N=32 output channels (2 halves),
// K=32 input channels per tap, 125 taps -> 250 WMMAs. Weights staged in LDS
// per kd-slice as fp16 [tap][co][ci]. Epilogue accumulates per-channel
// sum/sumsq (instance-norm stats) via LDS + global fp32 atomics.
// ---------------------------------------------------------------------------
__global__ __launch_bounds__(CONV_BLOCK)
void conv5_wmma_kernel(const _Float16* __restrict__ act,
                       const float* __restrict__ wgt,   // [CoutReal][CinReal][125] fp32
                       float* __restrict__ out,         // NDHWC fp32, C=32
                       float* __restrict__ stats,       // [32] sum, [32] sumsq
                       int CinReal, int CoutReal)
{
  __shared__ _Float16 sW[TAP_CHUNK * CCH * CCH];   // 25*32*32 f16 = 51200 B
  __shared__ float sSum[CCH], sSq[CCH];

  const int tid  = threadIdx.x;
  const int lane = tid & 31;
  const int wave = tid >> 5;

  const int tile = blockIdx.x * WAVES_PER_BLOCK + wave;  // 16-voxel tile
  const int v0 = tile * 16;
  const int w0 = v0 & 63;
  const int h0 = (v0 >> 6) & 63;
  const int d0 = v0 >> 12;

  const int mrow   = lane & 15;   // M voxel index base == N column index
  const int hiHalf = lane >> 4;   // 0: lanes 0-15, 1: lanes 16-31
  // A layout (16-bit 16x32): lane<16 holds ci {0..15},
  //                          lane>=16 holds ci {16..31}
  const int ciByte = hiHalf * 32;

  if (tid < CCH) { sSum[tid] = 0.f; sSq[tid] = 0.f; }

  v8f acc0 = {};   // output channels 0..15
  v8f acc1 = {};   // output channels 16..31

  for (int kd = 0; kd < 5; ++kd) {
    __syncthreads();   // protect previous chunk reads (and sSum init)
    // stage kd-slice weights into LDS as fp16 [tap][co][ci], zero-padded
    for (int e = tid; e < TAP_CHUNK * CCH * CCH; e += CONV_BLOCK) {
      int tap = e >> 10;          // / (32*32)
      int rem = e & 1023;
      int co  = rem >> 5;
      int ci  = rem & 31;
      float wv = 0.f;
      if (co < CoutReal && ci < CinReal)
        wv = wgt[(co * CinReal + ci) * TAPS + (kd * TAP_CHUNK + tap)];
      sW[e] = (_Float16)wv;
    }
    __syncthreads();

    const int dd = (d0 + kd - 2) & 63;
#pragma unroll
    for (int kh = 0; kh < 5; ++kh) {
      const int hh = (h0 + kh - 2) & 63;
      const int rowBase = ((dd << 6) + hh) << 6;   // voxel id of row start
#pragma unroll
      for (int kw = 0; kw < 5; ++kw) {
        const int ww = (w0 + mrow + kw - 2) & 63;  // circular pad=2
        const char* src = (const char*)(act + (size_t)(rowBase + ww) * CCH);
        union { v16h v; v8h h[2]; } a;
        a.h[0] = *(const v8h*)(src + ciByte);        // ci base .. +7
        a.h[1] = *(const v8h*)(src + ciByte + 16);   // ci base+8 .. +15

        const int tap = kh * 5 + kw;
        const _Float16* wb = &sW[tap * CCH * CCH];
        // B layout used here (same k order as A) (16-bit 32x16): lane<16 -> col N=lane, K=0..15;
        //                              lane>=16 -> col N=lane-16, K=16..31
        const v16h b0 = *(const v16h*)(wb + mrow * CCH + hiHalf * 16);
        const v16h b1 = *(const v16h*)(wb + (16 + mrow) * CCH + hiHalf * 16);

        acc0 = __builtin_amdgcn_wmma_f32_16x16x32_f16(
                   false, a.v, false, b0, (short)0, acc0, false, false);
        acc1 = __builtin_amdgcn_wmma_f32_16x16x32_f16(
                   false, a.v, false, b1, (short)0, acc1, false, false);
      }
    }
  }

  // D layout (32-bit 16x16): lane<16 -> M=r,  N=lane; lane>=16 -> M=8+r, N=lane-16
  float s0 = 0.f, q0 = 0.f, s1 = 0.f, q1 = 0.f;
#pragma unroll
  for (int r = 0; r < 8; ++r) {
    const int m = hiHalf * 8 + r;
    const size_t o = (size_t)(v0 + m) * CCH;
    float x0 = acc0[r];
    float x1 = acc1[r];
    out[o + mrow]      = x0;
    out[o + 16 + mrow] = x1;
    s0 += x0; q0 += x0 * x0;
    s1 += x1; q1 += x1 * x1;
  }
  atomicAdd(&sSum[mrow],      s0);
  atomicAdd(&sSq [mrow],      q0);
  atomicAdd(&sSum[16 + mrow], s1);
  atomicAdd(&sSq [16 + mrow], q1);
  __syncthreads();
  if (tid < CCH) {
    atomicAdd(&stats[tid],       sSum[tid]);
    atomicAdd(&stats[CCH + tid], sSq[tid]);
  }
}

// ---------------------------------------------------------------------------
// Instance-norm epilogue. modes:
//  0: h = leaky(norm(x));            write h (fp32) + act (fp16)
//  1: t = leaky(norm(x));            write act (fp16)
//  2: h = leaky(norm(x) + h_in);     write h (fp32) + act (fp16)
//  3: u = norm(x);                   write u (fp32, 3 channels compact)
// Biases are exactly absorbed by instnorm (constant channel shift), so omitted.
// ---------------------------------------------------------------------------
__global__ void norm_kernel(const float* __restrict__ conv,
                            const float* __restrict__ stats,
                            const float* __restrict__ hin,
                            float* __restrict__ hout,
                            _Float16* __restrict__ actOut,
                            float* __restrict__ uOut,
                            int mode)
{
  int idx = blockIdx.x * blockDim.x + threadIdx.x;   // over N3*CCH
  if (idx >= N3 * CCH) return;
  int c = idx & (CCH - 1);
  const float invN = 1.f / (float)N3;
  float m   = stats[c] * invN;
  float var = stats[CCH + c] * invN - m * m;
  float inv = rsqrtf(var + 1e-5f);
  float x = (conv[idx] - m) * inv;
  if (mode == 2) x += hin[idx];
  if (mode != 3) x = (x > 0.f) ? x : 0.01f * x;      // leaky_relu(0.01)
  if (mode == 0 || mode == 2) hout[idx] = x;
  if (mode != 3) {
    actOut[idx] = (_Float16)x;
  } else {
    int v = idx >> 5;
    if (c < 3) uOut[v * 3 + c] = x;
  }
}

// ---------------------------------------------------------------------------
// Fused strain + tensor rotation + stress:
//   eps = sym(grad u) (central diff, circular) + E_macro
//   sigma = alpha^T * (C0 : (alpha * eps * alpha^T)) * alpha
// which equals einsum(C4, eps) with C4 = R4(alpha) C0, without materializing C4.
// ---------------------------------------------------------------------------
__global__ void sigma_kernel(const float* __restrict__ quat,  // NCDHW, 4 ch
                             const float* __restrict__ u,     // [v][3] fp32
                             const float* __restrict__ C0g,   // 81
                             const float* __restrict__ Eg,    // 9
                             float* __restrict__ out)         // [3][3][N3]
{
  __shared__ float sC0[81];
  __shared__ float sE[9];
  int tid = threadIdx.x;
  if (tid < 81) sC0[tid] = C0g[tid];
  if (tid < 9)  sE[tid]  = Eg[tid];
  __syncthreads();

  int v = blockIdx.x * blockDim.x + tid;
  if (v >= N3) return;
  int w = v & 63, h = (v >> 6) & 63, d = v >> 12;

  float q0 = quat[v], q1 = quat[N3 + v], q2 = quat[2 * N3 + v], q3 = quat[3 * N3 + v];
  const float P = -1.f;
  float qbar = q0 * q0 - (q1 * q1 + q2 * q2 + q3 * q3);
  float A[3][3];
  A[0][0] = qbar + 2.f * q1 * q1;
  A[0][1] = 2.f * (q1 * q2 - P * q0 * q3);
  A[0][2] = 2.f * (q1 * q3 + P * q0 * q2);
  A[1][0] = 2.f * (q1 * q2 + P * q0 * q3);
  A[1][1] = qbar + 2.f * q2 * q2;
  A[1][2] = 2.f * (q2 * q3 - P * q0 * q1);
  A[2][0] = 2.f * (q1 * q3 - P * q0 * q2);
  A[2][1] = 2.f * (q2 * q3 + P * q0 * q1);
  A[2][2] = qbar + 2.f * q3 * q3;
  float al[3][3];                 // alpha = A^T  (swapaxes in reference)
#pragma unroll
  for (int m = 0; m < 3; ++m)
#pragma unroll
    for (int i = 0; i < 3; ++i) al[m][i] = A[i][m];

  // central differences with circular wrap: du[i][j] = d u_i / d x_j
  int dp = (d + 1) & 63, dm = (d - 1) & 63;
  int hp = (h + 1) & 63, hm = (h - 1) & 63;
  int wp = (w + 1) & 63, wm = (w - 1) & 63;
  int vDp = ((dp << 6) + h) * 64 + w, vDm = ((dm << 6) + h) * 64 + w;
  int vHp = ((d << 6) + hp) * 64 + w, vHm = ((d << 6) + hm) * 64 + w;
  int vWp = ((d << 6) + h) * 64 + wp, vWm = ((d << 6) + h) * 64 + wm;
  float du[3][3];
#pragma unroll
  for (int i = 0; i < 3; ++i) {
    du[i][0] = 0.5f * (u[vDp * 3 + i] - u[vDm * 3 + i]);
    du[i][1] = 0.5f * (u[vHp * 3 + i] - u[vHm * 3 + i]);
    du[i][2] = 0.5f * (u[vWp * 3 + i] - u[vWm * 3 + i]);
  }
  float eps[3][3];
#pragma unroll
  for (int i = 0; i < 3; ++i)
#pragma unroll
    for (int j = 0; j < 3; ++j)
      eps[i][j] = 0.5f * (du[i][j] + du[j][i]) + sE[i * 3 + j];

  // eps' = al * eps * al^T
  float T[3][3], ep[3][3];
#pragma unroll
  for (int o = 0; o < 3; ++o)
#pragma unroll
    for (int l = 0; l < 3; ++l)
      T[o][l] = al[o][0] * eps[0][l] + al[o][1] * eps[1][l] + al[o][2] * eps[2][l];
#pragma unroll
  for (int o = 0; o < 3; ++o)
#pragma unroll
    for (int p = 0; p < 3; ++p)
      ep[o][p] = T[o][0] * al[p][0] + T[o][1] * al[p][1] + T[o][2] * al[p][2];

  // S[m][n] = C0[m][n][o][p] : ep[o][p]
  float S[3][3];
#pragma unroll
  for (int m = 0; m < 3; ++m)
#pragma unroll
    for (int n = 0; n < 3; ++n) {
      float s = 0.f;
#pragma unroll
      for (int o = 0; o < 3; ++o)
#pragma unroll
        for (int p = 0; p < 3; ++p)
          s += sC0[((m * 3 + n) * 3 + o) * 3 + p] * ep[o][p];
      S[m][n] = s;
    }

  // sigma[i][j] = al[m][i] al[n][j] S[m][n]
  float U[3][3];
#pragma unroll
  for (int m = 0; m < 3; ++m)
#pragma unroll
    for (int j = 0; j < 3; ++j)
      U[m][j] = S[m][0] * al[0][j] + S[m][1] * al[1][j] + S[m][2] * al[2][j];
#pragma unroll
  for (int i = 0; i < 3; ++i)
#pragma unroll
    for (int j = 0; j < 3; ++j) {
      float sg = al[0][i] * U[0][j] + al[1][i] * U[1][j] + al[2][i] * U[2][j];
      out[(i * 3 + j) * N3 + v] = sg;
    }
}

// ---------------------------------------------------------------------------
extern "C" void kernel_launch(void* const* d_in, const int* in_sizes, int n_in,
                              void* d_out, int out_size, void* d_ws, size_t ws_size,
                              hipStream_t stream) {
  const float* quat = (const float*)d_in[0];
  // d_in[1] = allkernels: fixed +/-0.5 central-difference stencil, hardcoded.
  const float* C0g = (const float*)d_in[2];
  const float* Eg  = (const float*)d_in[3];

  // Identify weights robustly by element count (biases are zero and cancel
  // exactly under the instance norms, so they are unused).
  const float* wIn = nullptr;
  const float* wOut = nullptr;
  const float* wBlk[8] = {};
  int nb = 0;
  for (int i = 4; i < n_in; ++i) {
    int s = in_sizes[i];
    if (s == 32 * 4 * 125)            wIn = (const float*)d_in[i];
    else if (s == 3 * 32 * 125)       wOut = (const float*)d_in[i];
    else if (s == 32 * 32 * 125 && nb < 8) wBlk[nb++] = (const float*)d_in[i];
  }
  if (!wIn || !wOut || nb < 8) return;

  // workspace layout
  char* ws = (char*)d_ws;
  _Float16* actA = (_Float16*)(ws);                       // 16 MB fp16 NDHWC
  _Float16* actB = (_Float16*)(ws + (((size_t)16) << 20));
  float* conv    = (float*)(ws + (((size_t)32) << 20));   // 32 MB fp32 NDHWC
  float* hbuf    = (float*)(ws + (((size_t)64) << 20));   // 32 MB fp32 NDHWC
  float* ubuf    = (float*)(ws + (((size_t)96) << 20));   // 3 MB  fp32 [v][3]
  float* stats   = (float*)(ws + (((size_t)100) << 20));  // 256 B

  const int NC = N3 * CCH;
  const int convGrid = (N3 / 16) / WAVES_PER_BLOCK;  // 1024 blocks

  pack_quat_kernel<<<NC / 256, 256, 0, stream>>>(quat, actA);

  auto runconv = [&](const _Float16* in, const float* w, int cin, int cout) {
    (void)hipMemsetAsync(stats, 0, 2 * CCH * sizeof(float), stream);
    conv5_wmma_kernel<<<convGrid, CONV_BLOCK, 0, stream>>>(in, w, conv, stats, cin, cout);
  };

  // h = leaky(instnorm(conv_in(quat)))
  runconv(actA, wIn, 4, 32);
  norm_kernel<<<NC / 256, 256, 0, stream>>>(conv, stats, nullptr, hbuf, actB, nullptr, 0);

  // 4 residual blocks
  for (int r = 0; r < 4; ++r) {
    runconv(actB, wBlk[2 * r + 0], 32, 32);
    norm_kernel<<<NC / 256, 256, 0, stream>>>(conv, stats, nullptr, nullptr, actA, nullptr, 1);
    runconv(actA, wBlk[2 * r + 1], 32, 32);
    norm_kernel<<<NC / 256, 256, 0, stream>>>(conv, stats, hbuf, hbuf, actB, nullptr, 2);
  }

  // u = instnorm(conv_out(h))
  runconv(actB, wOut, 32, 3);
  norm_kernel<<<NC / 256, 256, 0, stream>>>(conv, stats, nullptr, nullptr, nullptr, ubuf, 3);

  // sigma = C4(alpha) : (sym(grad u) + E_macro)
  sigma_kernel<<<N3 / 256, 256, 0, stream>>>(quat, ubuf, C0g, Eg, (float*)d_out);
}
